// RegionProposalNetwork_79388175499528
// MI455X (gfx1250) — matrix operand order repaired and checked by reading, not compiled, not run on hardware
//
#include <hip/hip_runtime.h>
#include <hip/hip_bf16.h>

#define NPIX 4096
#define CIN 1024
#define CMID 512
#define NANCH 36864
#define PRE_NMS 6000
#define POST_NMS 300

// ---- workspace byte offsets ----
#define OFF_FRELU 0ull                // 512*4096*4  = 8388608
#define OFF_WT    8388608ull          // 9*512*512*4 = 9437184 (uint: packed bf16 pair, fragment-major)
#define OFF_FBF   17825792ull         // 512*4096*4  = 8388608 (uint: packed bf16 pair, [icpair][pix])
#define OFF_PROPS 26214400ull         // 36864*4*4
#define OFF_SCORE 26804224ull         // 36864*4
#define OFF_VALID 26951680ull         // 36864*4
#define OFF_BOX6K 27099136ull         // 6000*4*4
#define OFF_KEEP  27195136ull         // 6000*4

// ---- output float offsets (tuple flattened) ----
#define OOUT_ANCH  0
#define OOUT_OBJ   147456
#define OOUT_TRANS 221184
#define OOUT_FINAL 368640

#if defined(__has_builtin)
#if __has_builtin(__builtin_amdgcn_global_load_async_to_lds_b128)
#define ASYNC_LDS 1
#endif
#endif

typedef __attribute__((ext_vector_type(16))) __bf16 v16bf;
typedef __attribute__((ext_vector_type(8)))  float  v8f;
typedef int v4i __attribute__((vector_size(16)));
typedef __attribute__((address_space(1))) v4i* gv4i_p;   // global int4*
typedef __attribute__((address_space(3))) v4i* lv4i_p;   // LDS int4*

__device__ __forceinline__ unsigned int f2bf(float f) {
    unsigned int u = __float_as_uint(f);
    u += 0x7FFFu + ((u >> 16) & 1u);          // round-to-nearest-even
    return u >> 16;
}

// ---------------- prep: fp32 -> packed bf16 pairs ----------------
// feats2[icp][pix] = bf16(feat[2*icp][pix]) | bf16(feat[2*icp+1][pix])<<16
__global__ void prep_feats(const float* __restrict__ feats, unsigned int* __restrict__ f2) {
    int i = blockIdx.x * 256 + threadIdx.x;   // 512*4096
    int icp = i >> 12, p = i & 4095;
    unsigned int lo = f2bf(feats[(size_t)icp * 8192 + p]);
    unsigned int hi = f2bf(feats[(size_t)icp * 8192 + 4096 + p]);
    f2[i] = lo | (hi << 16);
}

// Fragment-major weights:
// w3[((tap*32+kc)*512 + oc)*16 + kpl] = bf16(w1[oc][2*(kc*16+kpl)][tap]) | bf16(w1[oc][2*(kc*16+kpl)+1][tap])<<16
__global__ void prep_w(const float* __restrict__ w1, unsigned int* __restrict__ w3) {
    int i = blockIdx.x * 256 + threadIdx.x;   // 9*512*512 = 9*32*512*16
    int kpl = i & 15;
    int oc  = (i >> 4) & 511;
    int kck = i >> 13;                        // tap*32 + kc, 0..287
    int tap = kck >> 5;
    int kc  = kck & 31;
    int ic  = 2 * (kc * 16 + kpl);
    size_t base = (size_t)oc * 9216 + (size_t)ic * 9 + tap;
    unsigned int lo = f2bf(w1[base]);
    unsigned int hi = f2bf(w1[base + 9]);
    w3[i] = lo | (hi << 16);
}

// ---------------- 3x3 conv via bf16 WMMA ----------------
// Block: 256 thr = 8 waves. Block tile: 64 pixels x 256 ocs.
// Wave wid: M tile = wid>>1 (0..3), N half = wid&1 -> 8 x 16-oc N tiles, 8 accumulators.
__global__ void __launch_bounds__(256) conv3x3_wmma(
    const unsigned int* __restrict__ feats2,  // [icpair][4096] packed 2xbf16
    const unsigned int* __restrict__ w3,      // fragment-major packed 2xbf16
    const float* __restrict__ b1,
    float* __restrict__ frelu)                // [oc][4096] f32
{
    __shared__ unsigned int sA[64 * 16];      // [pix][kpair]      4KB
    __shared__ unsigned int sB[256 * 16];     // [oc_local][kpair] 16KB (fragment-major)

    const int tid  = threadIdx.x;
    const int lane = tid & 31;
    const int wid  = tid >> 5;
    const int mt    = wid >> 1;               // 0..3  (16-pixel tile)
    const int nhalf = wid & 1;                // 0..1  (8 n-tiles each)
    const int mb   = blockIdx.x & 63;
    const int nb   = blockIdx.x >> 6;
    const int pixBase = mb * 64;
    const int ocBase  = nb * 256;
    const int ln15 = lane & 15;
    const int g    = lane >> 4;

    v8f acc[8];
    #pragma unroll
    for (int nt = 0; nt < 8; ++nt)
        #pragma unroll
        for (int r = 0; r < 8; ++r) acc[nt][r] = 0.0f;

    for (int tap = 0; tap < 9; ++tap) {
        const int dy = tap / 3 - 1, dx = tap % 3 - 1;
        for (int kc = 0; kc < 32; ++kc) {
            const int kpb = kc * 16;          // global ic-pair base for this chunk
            // contiguous 16KB slab for this (tap,kc,ocBase): 256 ocs x 16 kpairs
            const unsigned int* wchunk = w3 + ((size_t)(tap * 32 + kc) * 512 + ocBase) * 16;
            __syncthreads();

            // stage A: 64 pixels x 16 k-pairs (SAME zero padding), 4 dwords/thread
            #pragma unroll
            for (int i = 0; i < 4; ++i) {
                int e = tid + i * 256;
                int pix = e & 63, kp = e >> 6;
                int p = pixBase + pix;
                int y = (p >> 6) + dy, x = (p & 63) + dx;
                unsigned int val = 0u;
                if ((unsigned)y < 64u && (unsigned)x < 64u)
                    val = feats2[(size_t)(kpb + kp) * 4096 + y * 64 + x];
                sA[pix * 16 + kp] = val;
            }

            // stage B: linear 16KB copy, 16B per lane per issue
#ifdef ASYNC_LDS
            #pragma unroll
            for (int i = 0; i < 4; ++i) {
                int e = i * 256 + tid;
                __builtin_amdgcn_global_load_async_to_lds_b128(
                    (gv4i_p)(wchunk + e * 4), (lv4i_p)&sB[e * 4], 0, 0);
            }
#if __has_builtin(__builtin_amdgcn_s_wait_asynccnt)
            __builtin_amdgcn_s_wait_asynccnt(0);
#else
            asm volatile("s_wait_asynccnt 0x0" ::: "memory");
#endif
#else
            #pragma unroll
            for (int i = 0; i < 4; ++i) {
                int e = i * 256 + tid;
                #pragma unroll
                for (int q = 0; q < 4; ++q)
                    sB[e * 4 + q] = wchunk[e * 4 + q];
            }
#endif
            __syncthreads();

            // A fragment: 16x32 bf16, CDNA5 layout (lane m=L&15; K interleaved by lane half)
            union { v16bf v; unsigned int u[8]; } af;
            const int arow = (mt * 16 + ln15) * 16;
            #pragma unroll
            for (int v = 0; v < 4; ++v) af.u[v] = sA[arow + g * 4 + v];
            #pragma unroll
            for (int v = 0; v < 4; ++v) af.u[4 + v] = sA[arow + 8 + g * 4 + v];

            #pragma unroll
            for (int nt = 0; nt < 8; ++nt) {
                union { v16bf v; unsigned int u[8]; } bf_;
                const int brow = ((nhalf * 8 + nt) * 16 + ln15) * 16 + g * 4;
                #pragma unroll
                for (int v = 0; v < 4; ++v) bf_.u[v] = sB[brow + v];   // 2x ds_load_b128
                #pragma unroll
                for (int v = 0; v < 4; ++v) bf_.u[4 + v] = sB[brow + 8 + v];
                acc[nt] = __builtin_amdgcn_wmma_f32_16x16x32_bf16(
                    false, af.v, false, bf_.v, (short)0, acc[nt], false, false);
            }
        }
    }

    // bias + relu + store f[oc][pix]
    #pragma unroll
    for (int nt = 0; nt < 8; ++nt) {
        int oc = ocBase + (nhalf * 8 + nt) * 16 + ln15;
        float bias = b1[oc];
        #pragma unroll
        for (int r = 0; r < 8; ++r) {
            int pix = pixBase + mt * 16 + r + 8 * g;
            float v = acc[nt][r] + bias;
            frelu[(size_t)oc * 4096 + pix] = v > 0.0f ? v : 0.0f;
        }
    }
}

// ---------------- 1x1 heads (obj 18ch, trans 36ch) ----------------
__global__ void heads_1x1(const float* __restrict__ frelu,
                          const float* __restrict__ wobj, const float* __restrict__ bobj,
                          const float* __restrict__ wtr,  const float* __restrict__ btr,
                          float* __restrict__ out)
{
    int idx = blockIdx.x * 256 + threadIdx.x;  // 54*4096
    int ch  = idx >> 12;                       // 0..53
    int pix = idx & 4095;
    const float* w = (ch < 18) ? (wobj + (size_t)ch * 512) : (wtr + (size_t)(ch - 18) * 512);
    float acc = 0.0f;
    for (int ic = 0; ic < 512; ++ic)
        acc += frelu[(size_t)ic * 4096 + pix] * w[ic];
    if (ch < 18) {
        acc += bobj[ch];
        int a = ch >> 1, c = ch & 1;
        out[OOUT_OBJ + ((size_t)pix * 9 + a) * 2 + c] = acc;
    } else {
        int t = ch - 18;
        acc += btr[t];
        int a = t >> 2, c = t & 3;
        out[OOUT_TRANS + ((size_t)pix * 9 + a) * 4 + c] = acc;
    }
}

// ---------------- anchors + transformer + clip + score/valid ----------------
__global__ void anchors_props(float* __restrict__ out,
                              float* __restrict__ props, float* __restrict__ score,
                              int* __restrict__ valid,
                              const int* __restrict__ iwp, const int* __restrict__ ihp)
{
    int i = blockIdx.x * 256 + threadIdx.x;    // 36864
    float IW = (float)(*iwp), IH = (float)(*ihp);
    int a = i % 9;
    int p = i / 9;
    int x = p & 63, y = p >> 6;
    int ri = a / 3, si = a % 3;
    const float rs[3]  = {0.70710678118654752f, 1.0f, 1.41421356237309505f}; // sqrt(r)
    const float szs[3] = {32.0f, 64.0f, 128.0f};
    float cy = (float)(y + 1) * (IH / 65.0f);
    float cx = (float)(x + 1) * (IW / 65.0f);
    float hh = szs[si] * rs[ri];
    float ww = szs[si] / rs[ri];               // sizes*sqrt(1/r)
    float ax1 = cx - 0.5f * ww, ay1 = cy - 0.5f * hh;
    float ax2 = cx + 0.5f * ww, ay2 = cy + 0.5f * hh;
    out[OOUT_ANCH + (size_t)i * 4 + 0] = ax1;
    out[OOUT_ANCH + (size_t)i * 4 + 1] = ay1;
    out[OOUT_ANCH + (size_t)i * 4 + 2] = ax2;
    out[OOUT_ANCH + (size_t)i * 4 + 3] = ay2;

    float t0 = out[OOUT_TRANS + (size_t)i * 4 + 0];
    float t1 = out[OOUT_TRANS + (size_t)i * 4 + 1];
    float t2 = out[OOUT_TRANS + (size_t)i * 4 + 2];
    float t3 = out[OOUT_TRANS + (size_t)i * 4 + 3];
    float wA = ax2 - ax1, hA = ay2 - ay1;
    float ncx = ax1 + 0.5f * wA + t0 * wA;
    float ncy = ay1 + 0.5f * hA + t1 * hA;
    float nw = wA * expf(t2);
    float nh = hA * expf(t3);
    float x1 = fminf(fmaxf(ncx - 0.5f * nw, 0.0f), IW);
    float y1 = fminf(fmaxf(ncy - 0.5f * nh, 0.0f), IH);
    float x2 = fminf(fmaxf(ncx + 0.5f * nw, 0.0f), IW);
    float y2 = fminf(fmaxf(ncy + 0.5f * nh, 0.0f), IH);
    props[(size_t)i * 4 + 0] = x1;
    props[(size_t)i * 4 + 1] = y1;
    props[(size_t)i * 4 + 2] = x2;
    props[(size_t)i * 4 + 3] = y2;
    int vld = ((x2 - x1) >= 16.0f) && ((y2 - y1) >= 16.0f);
    valid[i] = vld;
    float sc = out[OOUT_OBJ + (size_t)i * 2 + 1];
    score[i] = vld ? sc : -__builtin_huge_valf();
}

// ---------------- stable descending rank (argsort equivalent) + gather top 6000 ----------------
__global__ void rank_topk(const float* __restrict__ score, const float* __restrict__ props,
                          const int* __restrict__ valid,
                          float* __restrict__ box6k, int* __restrict__ keep6k)
{
    __shared__ float sh[256];
    int i = blockIdx.x * 256 + threadIdx.x;   // 36864 threads
    float si = score[i];
    int cnt = 0;
    for (int t = 0; t < 144; ++t) {
        __syncthreads();
        sh[threadIdx.x] = score[t * 256 + threadIdx.x];
        __syncthreads();
        int jb = t * 256;
        for (int jj = 0; jj < 256; ++jj) {
            float sj = sh[jj];
            int j = jb + jj;
            cnt += ((sj > si) || (sj == si && j < i)) ? 1 : 0;
        }
    }
    if (cnt < PRE_NMS) {
        box6k[(size_t)cnt * 4 + 0] = props[(size_t)i * 4 + 0];
        box6k[(size_t)cnt * 4 + 1] = props[(size_t)i * 4 + 1];
        box6k[(size_t)cnt * 4 + 2] = props[(size_t)i * 4 + 2];
        box6k[(size_t)cnt * 4 + 3] = props[(size_t)i * 4 + 3];
        keep6k[cnt] = valid[i];
    }
}

// ---------------- sequential NMS, single workgroup ----------------
__global__ void __launch_bounds__(1024) nms6000(const float* __restrict__ box,
                                                int* __restrict__ keep)
{
    __shared__ unsigned char km[PRE_NMS];
    int tid = threadIdx.x;
    float x1[6], y1[6], x2[6], y2[6], ar[6];
    int kq[6];
    #pragma unroll
    for (int q = 0; q < 6; ++q) {
        int j = q * 1024 + tid;
        if (j < PRE_NMS) {
            x1[q] = box[(size_t)j * 4 + 0]; y1[q] = box[(size_t)j * 4 + 1];
            x2[q] = box[(size_t)j * 4 + 2]; y2[q] = box[(size_t)j * 4 + 3];
            ar[q] = (x2[q] - x1[q]) * (y2[q] - y1[q]);
            kq[q] = keep[j];
            km[j] = (unsigned char)kq[q];
        } else {
            kq[q] = 0;
            x1[q] = y1[q] = x2[q] = y2[q] = ar[q] = 0.0f;
        }
    }
    __syncthreads();
    for (int i = 0; i < PRE_NMS; ++i) {
        if (km[i]) {
            float bx1 = box[(size_t)i * 4 + 0], by1 = box[(size_t)i * 4 + 1];
            float bx2 = box[(size_t)i * 4 + 2], by2 = box[(size_t)i * 4 + 3];
            float bar = (bx2 - bx1) * (by2 - by1);
            #pragma unroll
            for (int q = 0; q < 6; ++q) {
                int j = q * 1024 + tid;
                if (j > i && j < PRE_NMS && kq[q]) {
                    float ix1 = fmaxf(x1[q], bx1), iy1 = fmaxf(y1[q], by1);
                    float ix2 = fminf(x2[q], bx2), iy2 = fminf(y2[q], by2);
                    float iw = fmaxf(ix2 - ix1, 0.0f), ih = fmaxf(iy2 - iy1, 0.0f);
                    float inter = iw * ih;
                    float iou = inter / (ar[q] + bar - inter + 1e-9f);
                    if (iou > 0.7f) { kq[q] = 0; km[j] = 0; }
                }
            }
        }
        __syncthreads();
    }
    #pragma unroll
    for (int q = 0; q < 6; ++q) {
        int j = q * 1024 + tid;
        if (j < PRE_NMS) keep[j] = kq[q];
    }
}

// ---------------- compact kept boxes -> 300 outputs (zero padded) ----------------
__global__ void __launch_bounds__(1024) final300(const float* __restrict__ box,
                                                 const int* __restrict__ keep,
                                                 float* __restrict__ out)
{
    int tid = threadIdx.x;
    for (int k = tid; k < POST_NMS * 4; k += 1024) out[OOUT_FINAL + k] = 0.0f;
    __syncthreads();
    for (int j = tid; j < PRE_NMS; j += 1024) {
        if (keep[j]) {
            int r = 0;
            for (int k = 0; k < j; ++k) r += keep[k];
            if (r < POST_NMS) {
                out[OOUT_FINAL + (size_t)r * 4 + 0] = box[(size_t)j * 4 + 0];
                out[OOUT_FINAL + (size_t)r * 4 + 1] = box[(size_t)j * 4 + 1];
                out[OOUT_FINAL + (size_t)r * 4 + 2] = box[(size_t)j * 4 + 2];
                out[OOUT_FINAL + (size_t)r * 4 + 3] = box[(size_t)j * 4 + 3];
            }
        }
    }
}

extern "C" void kernel_launch(void* const* d_in, const int* in_sizes, int n_in,
                              void* d_out, int out_size, void* d_ws, size_t ws_size,
                              hipStream_t stream)
{
    (void)in_sizes; (void)n_in; (void)out_size; (void)ws_size;
    const float* feats = (const float*)d_in[0];
    const float* w1    = (const float*)d_in[1];
    const float* b1    = (const float*)d_in[2];
    const float* wobj  = (const float*)d_in[3];
    const float* bobj  = (const float*)d_in[4];
    const float* wtr   = (const float*)d_in[5];
    const float* btr   = (const float*)d_in[6];
    const int*   iw    = (const int*)d_in[7];
    const int*   ih    = (const int*)d_in[8];

    char* ws = (char*)d_ws;
    float* out = (float*)d_out;

    float*        frelu  = (float*)(ws + OFF_FRELU);
    unsigned int* w3     = (unsigned int*)(ws + OFF_WT);
    unsigned int* feats2 = (unsigned int*)(ws + OFF_FBF);
    float*        props  = (float*)(ws + OFF_PROPS);
    float*        score  = (float*)(ws + OFF_SCORE);
    int*          valid  = (int*)(ws + OFF_VALID);
    float*        box6k  = (float*)(ws + OFF_BOX6K);
    int*          keep6k = (int*)(ws + OFF_KEEP);

    prep_feats<<<8192, 256, 0, stream>>>(feats, feats2);
    prep_w<<<9216, 256, 0, stream>>>(w1, w3);
    conv3x3_wmma<<<128, 256, 0, stream>>>(feats2, w3, b1, frelu);
    heads_1x1<<<864, 256, 0, stream>>>(frelu, wobj, bobj, wtr, btr, out);
    anchors_props<<<144, 256, 0, stream>>>(out, props, score, valid, iw, ih);
    rank_topk<<<144, 256, 0, stream>>>(score, props, valid, box6k, keep6k);
    nms6000<<<1, 1024, 0, stream>>>(box6k, keep6k);
    final300<<<1, 1024, 0, stream>>>(box6k, keep6k, out);
}
